// GATModel_14422500180490
// MI455X (gfx1250) — compile-verified, matrix operand order checked by reading, not run on hardware
//
#include <hip/hip_runtime.h>
#include <hip/hip_bf16.h>

typedef __attribute__((ext_vector_type(16))) _Float16 v16h;
typedef __attribute__((ext_vector_type(8)))  _Float16 v8h;
typedef __attribute__((ext_vector_type(8)))  float    v8f;

// ---------------------------------------------------------------- utilities
__device__ __forceinline__ unsigned ordEnc(float f) {
    unsigned b = __float_as_uint(f);
    return (b & 0x80000000u) ? ~b : (b | 0x80000000u);
}
__device__ __forceinline__ float ordDec(unsigned u) {
    return __uint_as_float((u & 0x80000000u) ? (u & 0x7FFFFFFFu) : ~u);
}

__global__ void cvt_f16_kernel(const float* __restrict__ src, _Float16* __restrict__ dst, int n) {
    int i = blockIdx.x * blockDim.x + threadIdx.x;
    if (i < n) dst[i] = (_Float16)src[i];
}

// ---------------------------------------------------------------- WMMA GEMM
// C[M,Ncols] = A[M,K](f16,row-major) * B[K,Ncols](f16,row-major), f32 accum.
// Block: 256 threads = 8 waves; tile 16 rows x 128 cols; K stepped by 32.
// Double-buffered LDS staging through the CDNA5 async copy engine
// (GLOBAL_LOAD_ASYNC_TO_LDS_*, ASYNCcnt) instead of load+ds_store.
// M % 16 == 0, K % 32 == 0, Ncols % 128 == 0 (guaranteed by problem sizes).
__global__ __launch_bounds__(256) void gemm_f16_wmma(
    const _Float16* __restrict__ A, const _Float16* __restrict__ B,
    float* __restrict__ C, int M, int K, int Ncols)
{
    __shared__ _Float16 sA[2][16 * 32];    // 2 x 1 KB
    __shared__ _Float16 sB[2][32 * 128];   // 2 x 8 KB

    const int m0   = blockIdx.x * 16;
    const int n0   = blockIdx.y * 128;
    const int tid  = threadIdx.x;
    const int wave = tid >> 5;
    const int lane = tid & 31;
    const int r    = lane & 15;     // row (A) / col-in-tile (B,C)
    const int hi   = lane >> 4;     // which K half-group / row half (C)

    // per-thread staging assignment (uniform shape across block)
    const int ia   = tid * 2;              // A: 2 halves  (4B async copy)
    const int arow = ia >> 5;
    const int acol = ia & 31;
    const int ib   = tid * 16;             // B: 16 halves (2x 16B async copies)
    const int brow = ib >> 7;
    const int bcol = ib & 127;

    auto stage_async = [&](int buf, int kc) {
        unsigned lA = (unsigned)(size_t)&sA[buf][ia];
        const _Float16* gA = &A[(size_t)(m0 + arow) * K + kc + acol];
        asm volatile("global_load_async_to_lds_b32 %0, %1, off"
                     :: "v"(lA), "v"(gA) : "memory");
        unsigned lB = (unsigned)(size_t)&sB[buf][ib];
        const _Float16* gB = &B[(size_t)(kc + brow) * Ncols + n0 + bcol];
        asm volatile("global_load_async_to_lds_b128 %0, %1, off"
                     :: "v"(lB), "v"(gB) : "memory");
        asm volatile("global_load_async_to_lds_b128 %0, %1, off offset:16"
                     :: "v"(lB), "v"(gB) : "memory");
    };

    v8f acc = {};

    // prologue: stage chunk 0
    stage_async(0, 0);
    asm volatile("s_wait_asynccnt 0x0" ::: "memory");
    __syncthreads();

    int cur = 0;
    for (int kc = 0; kc < K; kc += 32) {
        // kick off next chunk's async copies into the other buffer
        if (kc + 32 < K) stage_async(cur ^ 1, kc + 32);

        // A fragment: lane(r,hi): halves 0..7 -> K = hi*8+0..7 ; 8..15 -> K = 16+hi*8+0..7
        v8h alo = *(const v8h*)&sA[cur][r * 32 + hi * 8];
        v8h ahi = *(const v8h*)&sA[cur][r * 32 + 16 + hi * 8];
        v16h af = __builtin_shufflevector(alo, ahi,
                    0, 1, 2, 3, 4, 5, 6, 7, 8, 9, 10, 11, 12, 13, 14, 15);

        // B fragment: lane covers column n = wave*16 + r, same K distribution
        v16h bf = {};
        const int ncol = (wave << 4) + r;
        #pragma unroll
        for (int j = 0; j < 8; ++j) {
            bf[j]     = sB[cur][(hi * 8 + j) * 128 + ncol];
            bf[8 + j] = sB[cur][(16 + hi * 8 + j) * 128 + ncol];
        }

        acc = __builtin_amdgcn_wmma_f32_16x16x32_f16(
                false, af, false, bf, (short)0, acc, false, false);

        // make next buffer's async copies visible to the whole block
        asm volatile("s_wait_asynccnt 0x0" ::: "memory");
        __syncthreads();
        cur ^= 1;
    }

    // C 16x16 f32 layout: VGPR v -> row = v + 8*hi, col = r
    const int ccol = n0 + (wave << 4) + r;
    #pragma unroll
    for (int v = 0; v < 8; ++v)
        C[(size_t)(m0 + hi * 8 + v) * Ncols + ccol] = acc[v];
}

// ----------------------------------------------------- attention projections
// a_s[n,h] = dot(h[n,h,:], att_src[h,:]) ; a_d likewise. One wave per (n,h), C=128.
__global__ __launch_bounds__(256) void att_dots_kernel(
    const float* __restrict__ h, const float* __restrict__ as_w,
    const float* __restrict__ ad_w, float* __restrict__ a_s,
    float* __restrict__ a_d, int N, int H)
{
    int wid  = (blockIdx.x * blockDim.x + threadIdx.x) >> 5;
    int lane = threadIdx.x & 31;
    if (wid >= N * H) return;
    int n  = wid / H;
    int hh = wid - n * H;
    int ld = H * 128;
    const float4* hv = (const float4*)&h[(size_t)n * ld + hh * 128];
    const float4* sw = (const float4*)&as_w[hh * 128];
    const float4* dw = (const float4*)&ad_w[hh * 128];
    float4 v = hv[lane], s = sw[lane], d = dw[lane];
    float vs = v.x * s.x + v.y * s.y + v.z * s.z + v.w * s.w;
    float vd = v.x * d.x + v.y * d.y + v.z * d.z + v.w * d.w;
    #pragma unroll
    for (int off = 16; off; off >>= 1) {
        vs += __shfl_xor(vs, off, 32);
        vd += __shfl_xor(vd, off, 32);
    }
    if (lane == 0) { a_s[wid] = vs; a_d[wid] = vd; }
}

// ------------------------------------------------------------- edge passes
__global__ void edge_logits_kernel(
    const int* __restrict__ ei, int E, int H,
    const float* __restrict__ a_s, const float* __restrict__ a_d,
    float* __restrict__ val, unsigned* __restrict__ mmax, int Etot)
{
    int idx = blockIdx.x * blockDim.x + threadIdx.x;
    if (idx >= Etot * H) return;
    int e  = idx / H;
    int hh = idx - e * H;
    int s, d;
    if (e < E) { s = ei[e]; d = ei[E + e]; } else { s = d = e - E; }
    float v = a_s[s * H + hh] + a_d[d * H + hh];
    v = v > 0.f ? v : 0.2f * v;                 // leaky_relu(0.2)
    val[idx] = v;
    atomicMax(&mmax[d * H + hh], ordEnc(v));
}

__global__ void edge_softmax_num_kernel(
    const int* __restrict__ ei, int E, int H,
    const unsigned* __restrict__ mmax, float* __restrict__ val,
    float* __restrict__ denom, int Etot)
{
    int idx = blockIdx.x * blockDim.x + threadIdx.x;
    if (idx >= Etot * H) return;
    int e  = idx / H;
    int hh = idx - e * H;
    int d  = (e < E) ? ei[E + e] : e - E;
    float mo = ordDec(mmax[d * H + hh]);
    float ex = expf(val[idx] - mo);
    val[idx] = ex;
    atomicAdd(&denom[d * H + hh], ex);
}

// one wave per edge; H*128 floats per edge, float4 per lane per head
__global__ __launch_bounds__(256) void edge_aggregate_kernel(
    const int* __restrict__ ei, int E,
    const float* __restrict__ hfeat, const float* __restrict__ val,
    const float* __restrict__ denom, float* __restrict__ out, int H, int Etot)
{
    int wid  = (blockIdx.x * 256 + threadIdx.x) >> 5;
    int lane = threadIdx.x & 31;
    if (wid >= Etot) return;
    int e = wid, s, d;
    if (e < E) { s = ei[e]; d = ei[E + e]; } else { s = d = e - E; }
    int HC = H * 128;
    const float4* hs = (const float4*)&hfeat[(size_t)s * HC];
    float* od = &out[(size_t)d * HC];
    for (int t = 0; t < H; ++t) {
        float alpha = val[e * H + t] / (denom[d * H + t] + 1e-16f);
        float4 v = hs[t * 32 + lane];
        int base = t * 128 + lane * 4;
        atomicAdd(&od[base + 0], v.x * alpha);
        atomicAdd(&od[base + 1], v.y * alpha);
        atomicAdd(&od[base + 2], v.z * alpha);
        atomicAdd(&od[base + 3], v.w * alpha);
    }
}

// ------------------------------------------------------------- pointwise
__global__ void bias_elu_to_f16_kernel(
    const float* __restrict__ in, const float* __restrict__ bias,
    _Float16* __restrict__ outh, int total, int ld)
{
    int i = blockIdx.x * blockDim.x + threadIdx.x;
    if (i >= total) return;
    float v = in[i] + bias[i % ld];
    v = v > 0.f ? v : (expf(v) - 1.f);
    outh[i] = (_Float16)v;
}

__global__ void bias_elu_f32_kernel(
    float* __restrict__ data, const float* __restrict__ bias, int total, int ld)
{
    int i = blockIdx.x * blockDim.x + threadIdx.x;
    if (i >= total) return;
    float v = data[i] + bias[i % ld];
    v = v > 0.f ? v : (expf(v) - 1.f);
    data[i] = v;
}

// ------------------------------------------------------------- pooling/head
__global__ __launch_bounds__(256) void pool_kernel(
    const float* __restrict__ h, const int* __restrict__ batch,
    float* __restrict__ sums, float* __restrict__ cnts, int N)
{
    int wid  = (blockIdx.x * 256 + threadIdx.x) >> 5;
    int lane = threadIdx.x & 31;
    if (wid >= N) return;
    int g = batch[wid];
    const float4* hv = (const float4*)&h[(size_t)wid * 128];
    float4 v = hv[lane];
    float* sg = &sums[g * 128 + lane * 4];
    atomicAdd(&sg[0], v.x);
    atomicAdd(&sg[1], v.y);
    atomicAdd(&sg[2], v.z);
    atomicAdd(&sg[3], v.w);
    if (lane == 0) atomicAdd(&cnts[g], 1.0f);
}

__global__ void final_kernel(
    const float* __restrict__ sums, const float* __restrict__ cnts,
    const float* __restrict__ lin_w, const float* __restrict__ lin_b,
    float* __restrict__ out, int G)
{
    int wid  = (blockIdx.x * blockDim.x + threadIdx.x) >> 5;
    int lane = threadIdx.x & 31;
    if (wid >= G) return;
    float c = cnts[wid];
    c = c > 1.f ? c : 1.f;
    const float4* sv = (const float4*)&sums[wid * 128];
    const float4* wv = (const float4*)lin_w;
    float4 s = sv[lane], w = wv[lane];
    float p = (s.x * w.x + s.y * w.y + s.z * w.z + s.w * w.w) / c;
    #pragma unroll
    for (int off = 16; off; off >>= 1) p += __shfl_xor(p, off, 32);
    if (lane == 0) out[wid] = p + lin_b[0];
}

// ---------------------------------------------------------------- launcher
extern "C" void kernel_launch(void* const* d_in, const int* in_sizes, int n_in,
                              void* d_out, int out_size, void* d_ws, size_t ws_size,
                              hipStream_t stream) {
    const float* x        = (const float*)d_in[0];
    const int*   ei       = (const int*)d_in[1];   // [2,E]: src then dst
    const int*   batch    = (const int*)d_in[2];
    const float* W1       = (const float*)d_in[3];
    const float* att_src1 = (const float*)d_in[4];
    const float* att_dst1 = (const float*)d_in[5];
    const float* b1       = (const float*)d_in[6];
    const float* W2       = (const float*)d_in[7];
    const float* att_src2 = (const float*)d_in[8];
    const float* att_dst2 = (const float*)d_in[9];
    const float* b2       = (const float*)d_in[10];
    const float* lin_w    = (const float*)d_in[11];
    const float* lin_b    = (const float*)d_in[12];
    float* out = (float*)d_out;

    const int N    = in_sizes[0] / 128;   // 30000
    const int E    = in_sizes[1] / 2;     // 480000
    const int Etot = E + N;               // self-loops appended
    const int G    = out_size;            // 64

    // workspace carving (256B aligned)
    char* base = (char*)d_ws;
    size_t off = 0;
    auto carve = [&](size_t bytes) -> void* {
        void* p = base + off;
        off += (bytes + 255) & ~(size_t)255;
        return p;
    };
    _Float16* xh    = (_Float16*)carve((size_t)N * 128 * 2);
    _Float16* W1h   = (_Float16*)carve((size_t)128 * 512 * 2);
    _Float16* W2h   = (_Float16*)carve((size_t)512 * 128 * 2);
    float*    h1    = (float*)carve((size_t)N * 512 * 4);
    float*    agg1  = (float*)carve((size_t)N * 512 * 4);
    _Float16* h1h   = (_Float16*)carve((size_t)N * 512 * 2);
    float*    a_s   = (float*)carve((size_t)N * 4 * 4);
    float*    a_d   = (float*)carve((size_t)N * 4 * 4);
    unsigned* mbuf  = (unsigned*)carve((size_t)N * 4 * 4);
    float*    denom = (float*)carve((size_t)N * 4 * 4);
    float*    val   = (float*)carve((size_t)Etot * 4 * 4);
    float*    sums  = (float*)carve((size_t)G * 128 * 4);
    float*    cnts  = (float*)carve((size_t)G * 4);
    // layer-2 feature buffers alias the (then dead) h1 region
    float* h2   = h1;                     // [N,128]
    float* agg2 = h1 + (size_t)N * 128;   // [N,128]

    const int T = 256;

    // f32 -> f16 conversions
    cvt_f16_kernel<<<(N * 128 + T - 1) / T, T, 0, stream>>>(x, xh, N * 128);
    cvt_f16_kernel<<<(128 * 512 + T - 1) / T, T, 0, stream>>>(W1, W1h, 128 * 512);
    cvt_f16_kernel<<<(512 * 128 + T - 1) / T, T, 0, stream>>>(W2, W2h, 512 * 128);

    // ---- layer 1 ----
    dim3 g1(N / 16, 512 / 128);
    gemm_f16_wmma<<<g1, T, 0, stream>>>(xh, W1h, h1, N, 128, 512);

    att_dots_kernel<<<(N * 4 + 7) / 8, T, 0, stream>>>(h1, att_src1, att_dst1, a_s, a_d, N, 4);

    hipMemsetAsync(mbuf, 0, (size_t)N * 4 * 4, stream);
    hipMemsetAsync(denom, 0, (size_t)N * 4 * 4, stream);
    edge_logits_kernel<<<(Etot * 4 + T - 1) / T, T, 0, stream>>>(ei, E, 4, a_s, a_d, val, mbuf, Etot);
    edge_softmax_num_kernel<<<(Etot * 4 + T - 1) / T, T, 0, stream>>>(ei, E, 4, mbuf, val, denom, Etot);

    hipMemsetAsync(agg1, 0, (size_t)N * 512 * 4, stream);
    edge_aggregate_kernel<<<(Etot + 7) / 8, T, 0, stream>>>(ei, E, h1, val, denom, agg1, 4, Etot);

    bias_elu_to_f16_kernel<<<(N * 512 + T - 1) / T, T, 0, stream>>>(agg1, b1, h1h, N * 512, 512);

    // ---- layer 2 ----
    dim3 g2(N / 16, 128 / 128);
    gemm_f16_wmma<<<g2, T, 0, stream>>>(h1h, W2h, h2, N, 512, 128);

    att_dots_kernel<<<(N + 7) / 8, T, 0, stream>>>(h2, att_src2, att_dst2, a_s, a_d, N, 1);

    hipMemsetAsync(mbuf, 0, (size_t)N * 4, stream);
    hipMemsetAsync(denom, 0, (size_t)N * 4, stream);
    edge_logits_kernel<<<(Etot + T - 1) / T, T, 0, stream>>>(ei, E, 1, a_s, a_d, val, mbuf, Etot);
    edge_softmax_num_kernel<<<(Etot + T - 1) / T, T, 0, stream>>>(ei, E, 1, mbuf, val, denom, Etot);

    hipMemsetAsync(agg2, 0, (size_t)N * 128 * 4, stream);
    edge_aggregate_kernel<<<(Etot + 7) / 8, T, 0, stream>>>(ei, E, h2, val, denom, agg2, 1, Etot);

    bias_elu_f32_kernel<<<(N * 128 + T - 1) / T, T, 0, stream>>>(agg2, b2, N * 128, 128);

    // ---- pool + linear head ----
    hipMemsetAsync(sums, 0, (size_t)G * 128 * 4, stream);
    hipMemsetAsync(cnts, 0, (size_t)G * 4, stream);
    pool_kernel<<<(N + 7) / 8, T, 0, stream>>>(agg2, batch, sums, cnts, N);
    final_kernel<<<(G * 32 + T - 1) / T, T, 0, stream>>>(sums, cnts, lin_w, lin_b, out, G);
}
